// KAN_Convolution_4466765988468
// MI455X (gfx1250) — matrix operand order repaired and checked by reading, not compiled, not run on hardware
//
#include <hip/hip_runtime.h>

typedef __attribute__((ext_vector_type(16))) _Float16 v16h;
typedef __attribute__((ext_vector_type(8)))  float    v8f;

namespace {
constexpr int C_IN = 16, H_IN = 128, W_IN = 128, B_N = 8;
constexpr int HO = 127, WO = 127;
constexpr int NKNOT = 12;
}

__device__ __forceinline__ float silu_fast(float x) {
  float e = __expf(-x);
  return x * __builtin_amdgcn_rcpf(1.0f + e);
}

// 8 cubic B-spline bases via Cox-de-Boor with precomputed reciprocal denominators.
__device__ __forceinline__ void bspline8(
    float x, const float (&g)[12],
    const float (&i1)[11], const float (&i2)[10], const float (&i3)[9],
    float (&out)[8]) {
  float b0[11];
#pragma unroll
  for (int i = 0; i < 11; ++i)
    b0[i] = (x >= g[i] && x < g[i + 1]) ? 1.0f : 0.0f;
  float b1[10];
#pragma unroll
  for (int i = 0; i < 10; ++i)
    b1[i] = (x - g[i]) * i1[i] * b0[i] + (g[i + 2] - x) * i1[i + 1] * b0[i + 1];
  float b2[9];
#pragma unroll
  for (int i = 0; i < 9; ++i)
    b2[i] = (x - g[i]) * i2[i] * b1[i] + (g[i + 3] - x) * i2[i + 1] * b1[i + 1];
#pragma unroll
  for (int i = 0; i < 8; ++i)
    out[i] = (x - g[i]) * i3[i] * b2[i] + (g[i + 4] - x) * i3[i + 1] * b2[i + 1];
}

__global__ __launch_bounds__(256, 1) void kan_conv2d_wmma(
    const float* __restrict__ x,
    const float* __restrict__ base_w,
    const float* __restrict__ spline_w,
    const float* __restrict__ scaler,
    const float* __restrict__ grid,
    float* __restrict__ out) {
  const int lane  = threadIdx.x & 31;
  const int wave  = threadIdx.x >> 5;
  const int row   = blockIdx.x;          // (b, ho)
  const int b     = row / HO;
  const int ho    = row - b * HO;
  const int p     = lane & 15;           // pixel within group
  const int upper = lane >> 4;           // 0: taps (0,0),(1,0); 1: taps (0,1),(1,1)
  const int wo    = wave * 16 + p;
  const int f1    = upper;               // first tap feature index
  const int f2    = upper + 2;           // second tap feature index
  int col = wo + upper;                  // x column for this lane (clamped, store masked)
  col = (col < W_IN - 1) ? col : (W_IN - 1);

  // Load this lane's two knot rows; precompute 1/(g[i+k]-g[i]) once per wave.
  // Denominators are knot spacings (O(1) constants) -> v_rcp_f32 accuracy is plenty,
  // and avoids ~60 full IEEE divide expansions of per-wave setup.
  float ga[12], gb[12];
#pragma unroll
  for (int i = 0; i < 12; ++i) {
    ga[i] = grid[f1 * NKNOT + i];
    gb[i] = grid[f2 * NKNOT + i];
  }
  float i1a[11], i2a[10], i3a[9];
  float i1b[11], i2b[10], i3b[9];
#pragma unroll
  for (int i = 0; i < 11; ++i) {
    i1a[i] = __builtin_amdgcn_rcpf(ga[i + 1] - ga[i]);
    i1b[i] = __builtin_amdgcn_rcpf(gb[i + 1] - gb[i]);
  }
#pragma unroll
  for (int i = 0; i < 10; ++i) {
    i2a[i] = __builtin_amdgcn_rcpf(ga[i + 2] - ga[i]);
    i2b[i] = __builtin_amdgcn_rcpf(gb[i + 2] - gb[i]);
  }
#pragma unroll
  for (int i = 0; i < 9; ++i) {
    i3a[i] = __builtin_amdgcn_rcpf(ga[i + 3] - ga[i]);
    i3b[i] = __builtin_amdgcn_rcpf(gb[i + 3] - gb[i]);
  }

  const float bw1 = base_w[f1];
  const float bw2 = base_w[f2];

  // B matrix: scaled spline weights replicated across all 16 columns.
  // Lanes 0-15 pack K0-7,K16-23; lanes 16-31 pack K8-15,K24-31.
  v16h bmat;
#pragma unroll
  for (int j = 0; j < 16; ++j) {
    const int k = (j & 8) * 2 + upper * 8 + (j & 7);
    bmat[j] = (_Float16)(spline_w[k] * scaler[k >> 3]);
  }

  v8f dacc = {0.f, 0.f, 0.f, 0.f, 0.f, 0.f, 0.f, 0.f};
  float bacc = 0.0f;

  const float* xp = x + (((size_t)b * C_IN) * H_IN + ho) * W_IN + col;
#pragma unroll 2
  for (int c = 0; c < C_IN; ++c) {
    const float xa = xp[0];        // (ho,   col)
    const float xb = xp[W_IN];     // (ho+1, col)
    xp += H_IN * W_IN;

    bacc += bw1 * silu_fast(xa) + bw2 * silu_fast(xb);

    float s1[8], s2[8];
    bspline8(xa, ga, i1a, i2a, i3a, s1);
    bspline8(xb, gb, i1b, i2b, i3b, s2);

    // A layout (f16 16x32): lane<16 -> K0-7,K16-23 ; lane>=16 -> K8-15,K24-31.
    v16h amat;
#pragma unroll
    for (int j = 0; j < 8; ++j) {
      amat[j]     = (_Float16)s1[j];
      amat[j + 8] = (_Float16)s2[j];
    }
    dacc = __builtin_amdgcn_wmma_f32_16x16x32_f16(
        false, amat, false, bmat, (short)0, dacc, false, false);
  }

  // Base path: combine the two lane-halves (taps 0,2 + taps 1,3) per pixel.
  bacc += __shfl_xor(bacc, 16, 32);

  // Spline path: D[m][n] identical across n. Lane l, VGPR r holds pixel r+8*upper.
  const int r = lane & 7;
  float v = dacc[0];
#pragma unroll
  for (int i = 1; i < 8; ++i) v = (r == i) ? dacc[i] : v;
  // lane l now holds pixel (l&7)+8*upper; route pixel l to lane l (l<16).
  const int src = (p < 8) ? p : (p + 8);
  const float spl = __shfl(v, src, 32);

  if (lane < 16 && wo < WO) {
    out[((size_t)b * HO + ho) * WO + wo] = spl + bacc;
  }
}

extern "C" void kernel_launch(void* const* d_in, const int* /*in_sizes*/, int /*n_in*/,
                              void* d_out, int /*out_size*/, void* /*d_ws*/, size_t /*ws_size*/,
                              hipStream_t stream) {
  const float* x  = (const float*)d_in[0];
  const float* bw = (const float*)d_in[1];
  const float* sw = (const float*)d_in[2];
  const float* sc = (const float*)d_in[3];
  const float* gr = (const float*)d_in[4];
  float* out = (float*)d_out;
  hipLaunchKernelGGL(kan_conv2d_wmma, dim3(B_N * HO), dim3(256), 0, stream,
                     x, bw, sw, sc, gr, out);
}
